// InputAttentionEncoder_62543313764534
// MI455X (gfx1250) — matrix operand order repaired and checked by reading, not compiled, not run on hardware
//
#include <hip/hip_runtime.h>
#include <hip/hip_bf16.h>
#include <math.h>

// Problem constants (from reference)
#define BB 512
#define TT 255
#define DD 128
#define HH 1024
#define G4H (4 * HH)

typedef __attribute__((ext_vector_type(16))) __bf16 v16bf;
typedef __attribute__((ext_vector_type(8)))  __bf16 v8bf;
typedef __attribute__((ext_vector_type(8)))  float  v8f;
typedef __attribute__((ext_vector_type(4)))  unsigned int u32x4;
typedef __attribute__((ext_vector_type(8)))  int i32x8;
typedef __attribute__((ext_vector_type(4)))  int i32x4;

__device__ __forceinline__ float sigmoidf_(float x) {
    return 1.0f / (1.0f + __expf(-x));
}

// Assemble a v16bf fragment from two 16B chunks (16B-aligned LDS reads ->
// ds_load_b128; avoids requiring 32B alignment on padded rows).
__device__ __forceinline__ v16bf frag16(const __bf16* p0, const __bf16* p1) {
    union { v16bf v; v8bf h[2]; } u;
    u.h[0] = *reinterpret_cast<const v8bf*>(p0);
    u.h[1] = *reinterpret_cast<const v8bf*>(p1);
    return u.v;
}

// ---------------------------------------------------------------------------
// TDM: issue one 2D tile load (tile = 32 K-elems x `rows`) into LDS.
// D# packing per CDNA5 ISA:
//   group0: [1:0]=count=1, [63:32]=lds_addr, [120:64]=global_addr, [127:126]=2
//   group1: data_size=2B, pad_enable, pad_interval=16 dwords (enc 3),
//           pad_amount=4 dwords (enc 3)  -> LDS row stride 80B (bank destagger),
//           tensor_dim0=ld, tensor_dim1=big, tile_dim0=32, tile_dim1=rows,
//           tensor_dim0_stride=ld
// This driver's toolchain exposes the 6-arg clang-23 builtin:
//   (u32x4 g0, i32x8 g1, i32x4 g2, i32x4 g3, i32x8, i32 cpol)
// ---------------------------------------------------------------------------
__device__ __forceinline__ void tdm_tile_load(unsigned lds_off, const void* gp,
                                              unsigned rows, unsigned ld_elems) {
    const unsigned long long ga = (unsigned long long)(size_t)gp;
    u32x4 g0;
    g0[0] = 1u;                                                  // count=1
    g0[1] = lds_off;                                             // LDS byte addr
    g0[2] = (unsigned)ga;                                        // addr[31:0]
    g0[3] = (unsigned)((ga >> 32) & 0x01FFFFFFull) | 0x80000000u; // addr[56:32]|type=2

    const unsigned td1 = 1u << 20;                               // huge dim1: no OOB
    i32x8 g1;
    g1[0] = (int)((1u << 16) | (1u << 20) | (3u << 22) | (3u << 25)); // dsz=2B,pad
    g1[1] = (int)((ld_elems & 0xFFFFu) << 16);                   // tensor_dim0 lo
    g1[2] = (int)(((ld_elems >> 16) & 0xFFFFu) | ((td1 & 0xFFFFu) << 16));
    g1[3] = (int)(((td1 >> 16) & 0xFFFFu) | (32u << 16));        // tile_dim0=32
    g1[4] = (int)(rows & 0xFFFFu);                               // tile_dim1; dim2=0
    g1[5] = (int)ld_elems;                                       // dim0_stride lo
    g1[6] = 0;
    g1[7] = 0;
    const i32x4 z4 = {0, 0, 0, 0};
    const i32x8 z8 = {0, 0, 0, 0, 0, 0, 0, 0};
    __builtin_amdgcn_tensor_load_to_lds(g0, g1, z4, z4, z8, 0);
}

// ---------------------------------------------------------------------------
// Kernel 1: time-invariant attention weights  a[b,d] = softmax_d(e_x[b,d])
// (softmax shift-invariance kills the recurrent h·wh + c·wc + b_attn term)
// ---------------------------------------------------------------------------
__global__ __launch_bounds__(DD)
void attn_softmax_kernel(const float* __restrict__ x,
                         const float* __restrict__ Wattn,
                         float* __restrict__ a) {
    const int b = blockIdx.x;
    const int d = threadIdx.x;
    const float* wx = Wattn + 2 * HH;
    const float* xb = x + (size_t)b * TT * DD + d;

    float s = 0.0f;
    #pragma unroll 4
    for (int t = 0; t < TT; ++t) s += xb[(size_t)t * DD] * wx[t];

    __shared__ float red[DD];
    red[d] = s;
    __syncthreads();
    for (int st = DD / 2; st > 0; st >>= 1) {
        if (d < st) red[d] = fmaxf(red[d], red[d + st]);
        __syncthreads();
    }
    const float mx = red[0];
    __syncthreads();
    const float e = __expf(s - mx);
    red[d] = e;
    __syncthreads();
    for (int st = DD / 2; st > 0; st >>= 1) {
        if (d < st) red[d] += red[d + st];
        __syncthreads();
    }
    a[(size_t)b * DD + d] = e / red[0];
}

// ---------------------------------------------------------------------------
// Kernel 2: input_weighted = a[:,None,:] * x   (fp32 out + bf16 staging)
// ---------------------------------------------------------------------------
__global__ __launch_bounds__(256)
void weight_input_kernel(const float* __restrict__ x,
                         const float* __restrict__ a,
                         float* __restrict__ out_w,
                         __bf16* __restrict__ wbf) {
    const size_t n = (size_t)BB * TT * DD;
    const size_t stride = (size_t)gridDim.x * blockDim.x;
    for (size_t i = (size_t)blockIdx.x * blockDim.x + threadIdx.x; i < n; i += stride) {
        const int d = (int)(i % DD);
        const int b = (int)(i / ((size_t)TT * DD));
        const float w = a[(size_t)b * DD + d] * x[i];
        out_w[i] = w;
        wbf[i] = (__bf16)w;
    }
}

// ---------------------------------------------------------------------------
// Kernel 3: convert weights to bf16, fuse biases, zero states
// ---------------------------------------------------------------------------
__global__ __launch_bounds__(256)
void prep_kernel(const float* __restrict__ W_ih, const float* __restrict__ W_hh,
                 const float* __restrict__ b_ih, const float* __restrict__ b_hh,
                 __bf16* __restrict__ Wihb, __bf16* __restrict__ Whhb,
                 float* __restrict__ bias,
                 __bf16* __restrict__ h0, float* __restrict__ cst) {
    const size_t n = (size_t)G4H * HH;
    const size_t stride = (size_t)gridDim.x * blockDim.x;
    for (size_t i = (size_t)blockIdx.x * blockDim.x + threadIdx.x; i < n; i += stride) {
        Whhb[i] = (__bf16)W_hh[i];
        if (i < (size_t)G4H * DD) Wihb[i] = (__bf16)W_ih[i];
        if (i < (size_t)G4H)      bias[i] = b_ih[i] + b_hh[i];
        if (i < (size_t)BB * HH) { h0[i] = (__bf16)0.0f; cst[i] = 0.0f; }
    }
}

// ---------------------------------------------------------------------------
// Kernel 4 (x255): one LSTM timestep.
// TDM double-buffered LDS staging of A tile (32x32) + 4 gate weight tiles
// (64x32 each); v_wmma_f32_16x16x32_bf16 compute; fused cell epilogue.
// Grid (16,16); block 256 = 8 waves; block patch = 32 batch x 64 hidden.
// ---------------------------------------------------------------------------
__global__ __launch_bounds__(256)
void lstm_step_kernel(const __bf16* __restrict__ wbf,   // (B,T,D) staged input
                      const __bf16* __restrict__ hin,   // (B,H) bf16
                      __bf16* __restrict__ hout,        // (B,H) bf16
                      float* __restrict__ cst,          // (B,H) fp32
                      const __bf16* __restrict__ Wihb,  // (4H,D) row-major
                      const __bf16* __restrict__ Whhb,  // (4H,H) row-major
                      const float* __restrict__ bias,   // (4H)
                      float* __restrict__ enc,          // (B,T,H) fp32 out
                      int t) {
    constexpr int ROWS = 40;                    // 32 data + 8 pad bf16 (80B rows)
    constexpr int AELEMS = 32 * ROWS;           // 1280
    constexpr int GATEELEMS = 64 * ROWS;        // 2560
    constexpr int BUFELEMS = AELEMS + 4 * GATEELEMS; // 11520 elems = 23040 B
    __shared__ __align__(128) __bf16 lds[2 * BUFELEMS];  // 46 KB

    const int lane = threadIdx.x & 31;
    const int wave = threadIdx.x >> 5;
    const int mt = wave >> 2;                   // 0..1 batch sub-tile
    const int ht = wave & 3;                    // 0..3 hidden sub-tile
    const int m0blk = blockIdx.x * 32;
    const int hcblk = blockIdx.y * 64;
    const int m0 = m0blk + mt * 16;
    const int hc = hcblk + ht * 16;

    // WMMA 16-bit fragment lane mapping (CDNA5 ISA):
    const int arow  = lane & 15;
    const int akoff = (lane >> 4) * 8;
    const int bcol  = lane & 15;
    const int bkoff = (lane >> 4) * 8;

    const unsigned ldsbase = (unsigned)(size_t)(&lds[0]);

    // Issue the 5 TDM descriptors for K-step j into LDS buffer b (wave 0 only).
    auto issue_tile = [&](int j, int b) {
        const bool s1 = j < 4;                  // segment 1: w_t @ W_ih^T (K=128)
        const int k0 = s1 ? (j * 32) : ((j - 4) * 32);
        const __bf16* Ap = s1 ? (wbf + (size_t)m0blk * (TT * DD) + (size_t)t * DD + k0)
                              : (hin + (size_t)m0blk * HH + k0);
        const unsigned Ald = s1 ? (unsigned)(TT * DD) : (unsigned)HH;
        const __bf16* Bw = s1 ? Wihb : Whhb;
        const unsigned Bld = s1 ? (unsigned)DD : (unsigned)HH;
        const unsigned base = ldsbase + (unsigned)b * (BUFELEMS * 2);
        tdm_tile_load(base, Ap, 32, Ald);
        #pragma unroll
        for (int g = 0; g < 4; ++g)
            tdm_tile_load(base + (unsigned)(AELEMS + g * GATEELEMS) * 2,
                          Bw + (size_t)(g * HH + hcblk) * Bld + k0, 64, Bld);
    };

    if (threadIdx.x < 32) issue_tile(0, 0);     // prologue: tile 0 -> buffer 0

    v8f acc0 = {}, acc1 = {}, acc2 = {}, acc3 = {};
    int buf = 0;
    #pragma unroll 1
    for (int iter = 0; iter < 36; ++iter) {
        if (threadIdx.x < 32) __builtin_amdgcn_s_wait_tensorcnt(0);
        __syncthreads();                        // tile `iter` ready in lds[buf]
        if (iter + 1 < 36 && threadIdx.x < 32)
            issue_tile(iter + 1, buf ^ 1);      // DMA overlaps the WMMAs below

        const __bf16* Lb = lds + (size_t)buf * BUFELEMS;
        const __bf16* ap = Lb + (mt * 16 + arow) * ROWS;
        const v16bf af = frag16(ap + akoff, ap + 16 + akoff);
        const __bf16* bp = Lb + AELEMS + (ht * 16 + bcol) * ROWS + bkoff;
        const v16bf b0 = frag16(bp, bp + 16);
        const v16bf b1 = frag16(bp + GATEELEMS, bp + GATEELEMS + 16);
        const v16bf b2 = frag16(bp + 2 * GATEELEMS, bp + 2 * GATEELEMS + 16);
        const v16bf b3 = frag16(bp + 3 * GATEELEMS, bp + 3 * GATEELEMS + 16);

        acc0 = __builtin_amdgcn_wmma_f32_16x16x32_bf16(false, af, false, b0, (short)0, acc0, false, false);
        acc1 = __builtin_amdgcn_wmma_f32_16x16x32_bf16(false, af, false, b1, (short)0, acc1, false, false);
        acc2 = __builtin_amdgcn_wmma_f32_16x16x32_bf16(false, af, false, b2, (short)0, acc2, false, false);
        acc3 = __builtin_amdgcn_wmma_f32_16x16x32_bf16(false, af, false, b3, (short)0, acc3, false, false);

        __syncthreads();                        // reads done before buf is reused
        buf ^= 1;
    }

    // ---- fused LSTM cell epilogue (gate order: i, f, g, o) ----
    // C/D layout: VGPR r -> M = r (lanes 0-15) or 8+r (lanes 16-31); N = lane&15.
    const int col = hc + (lane & 15);
    const int rbase = m0 + ((lane >> 4) ? 8 : 0);
    const float bi = bias[col];
    const float bf = bias[HH + col];
    const float bg = bias[2 * HH + col];
    const float bo = bias[3 * HH + col];
    #pragma unroll
    for (int r = 0; r < 8; ++r) {
        const int m = rbase + r;
        const size_t sidx = (size_t)m * HH + col;
        const float iv = sigmoidf_(acc0[r] + bi);
        const float fv = sigmoidf_(acc1[r] + bf);
        const float gv = tanhf(acc2[r] + bg);
        const float ov = sigmoidf_(acc3[r] + bo);
        const float cn = fv * cst[sidx] + iv * gv;
        cst[sidx] = cn;
        const float hn = ov * tanhf(cn);
        enc[(size_t)m * (TT * HH) + (size_t)t * HH + col] = hn;
        hout[sidx] = (__bf16)hn;
    }
}

// ---------------------------------------------------------------------------
extern "C" void kernel_launch(void* const* d_in, const int* in_sizes, int n_in,
                              void* d_out, int out_size, void* d_ws, size_t ws_size,
                              hipStream_t stream) {
    const float* x     = (const float*)d_in[0];  // (B,T,D)
    const float* Wattn = (const float*)d_in[1];  // (1, 2H+T)
    // d_in[2] = b_attn : mathematically irrelevant (softmax shift invariance)
    const float* W_ih  = (const float*)d_in[3];  // (4H, D)
    const float* W_hh  = (const float*)d_in[4];  // (4H, H)
    const float* b_ih  = (const float*)d_in[5];  // (4H)
    const float* b_hh  = (const float*)d_in[6];  // (4H)

    float* out_w = (float*)d_out;                          // (B,T,D)
    float* out_h = out_w + (size_t)BB * TT * DD;           // (B,T,H)

    // workspace carve-up (256B aligned)
    char* ws = (char*)d_ws;
    size_t off = 0;
    auto carve = [&](size_t bytes) -> char* {
        char* p = ws + off;
        off = (off + bytes + 255) & ~(size_t)255;
        return p;
    };
    float*  a_soft = (float*) carve((size_t)BB * DD * sizeof(float));
    __bf16* wbf    = (__bf16*)carve((size_t)BB * TT * DD * sizeof(__bf16));
    __bf16* Wihb   = (__bf16*)carve((size_t)G4H * DD * sizeof(__bf16));
    __bf16* Whhb   = (__bf16*)carve((size_t)G4H * HH * sizeof(__bf16));
    float*  bias   = (float*) carve((size_t)G4H * sizeof(float));
    __bf16* hbuf0  = (__bf16*)carve((size_t)BB * HH * sizeof(__bf16));
    __bf16* hbuf1  = (__bf16*)carve((size_t)BB * HH * sizeof(__bf16));
    float*  cst    = (float*) carve((size_t)BB * HH * sizeof(float));

    attn_softmax_kernel<<<BB, DD, 0, stream>>>(x, Wattn, a_soft);
    prep_kernel<<<4096, 256, 0, stream>>>(W_ih, W_hh, b_ih, b_hh,
                                          Wihb, Whhb, bias, hbuf0, cst);
    weight_input_kernel<<<8192, 256, 0, stream>>>(x, a_soft, out_w, wbf);

    __bf16* hb[2] = {hbuf0, hbuf1};
    for (int t = 0; t < TT; ++t) {
        lstm_step_kernel<<<dim3(BB / 32, HH / 64), 256, 0, stream>>>(
            wbf, hb[t & 1], hb[(t + 1) & 1], cst,
            Wihb, Whhb, bias, out_h, t);
    }
}
